// SwinTransformerBlockPost_2559800508440
// MI455X (gfx1250) — matrix operand order repaired and checked by reading, not compiled, not run on hardware
//
#include <hip/hip_runtime.h>
#include <hip/hip_bf16.h>
#include <math.h>

typedef __attribute__((ext_vector_type(16))) _Float16 v16h;
typedef __attribute__((ext_vector_type(8)))  _Float16 v8h;
typedef __attribute__((ext_vector_type(8)))  float    v8f;

// ---------------------------------------------------------------------------
// WMMA helpers (gfx1250, wave32). D = A(16x32,f16) * B(32x16,f16) + C(f32)
// A layout: lane<16 -> row=lane, k-chunks at k0+0..7 and k0+16..23
//           lane>=16 -> row=lane-16, chunks at k0+8..15 and k0+24..31
// B layout: lane<16 -> col=lane, k0+0..7 and k0+16..23; lane>=16 -> col=lane-16, k0+8..15 and k0+24..31
// D layout: elem v -> (row = m0 + v + 8*(lane>=16), col = n0 + (lane&15))
// ---------------------------------------------------------------------------
__device__ __forceinline__ v16h make_v16(v8h lo, v8h hi) {
  v16h r;
#pragma unroll
  for (int i = 0; i < 8; ++i) { r[i] = lo[i]; r[i + 8] = hi[i]; }
  return r;
}

// src points at element (row0, k0) of a row-major [rows x stride] f16 matrix.
__device__ __forceinline__ v16h frag_a(const _Float16* src, int stride, int lane) {
  const int l15 = lane & 15, lhi = (lane >> 4) & 1;
  const _Float16* p = src + (size_t)l15 * stride + lhi * 8;
  v8h lo = *(const v8h*)(p);
  v8h hi = *(const v8h*)(p + 16);
  return make_v16(lo, hi);
}

// src points at element (col0, k0) of a row-major [cols x stride] f16 matrix
// holding B transposed, i.e. Bt[n][k] == B[k][n].
__device__ __forceinline__ v16h frag_b(const _Float16* src, int stride, int lane) {
  const int l15 = lane & 15, lhi = (lane >> 4) & 1;
  const _Float16* p = src + (size_t)l15 * stride + lhi * 8;
  v8h lo = *(const v8h*)(p);
  v8h hi = *(const v8h*)(p + 16);
  return make_v16(lo, hi);
}

__device__ __forceinline__ v8f wmma32(v16h a, v16h b, v8f c) {
  return __builtin_amdgcn_wmma_f32_16x16x32_f16(false, a, false, b, (short)0, c,
                                                false, false);
}

// ---------------------------------------------------------------------------
// Constants
// ---------------------------------------------------------------------------
#define Bb    32
#define Hh    56
#define Ww    56
#define Cc    192
#define WSz   7
#define SHIFT 3
#define NHh   6
#define Ntok  49
#define HD    32
#define NWIN  64
#define Lseq  (Hh * Ww)          // 3136
#define ROWS  (Bb * Lseq)        // 100352
#define SCALEQ 0.17677669529663687f

// ---------------------------------------------------------------------------
// f32 -> f16 conversion
// ---------------------------------------------------------------------------
__global__ void cvt_kernel(const float* __restrict__ s, _Float16* __restrict__ d, int n) {
  int i = blockIdx.x * 256 + threadIdx.x;
  if (i < n) d[i] = (_Float16)s[i];
}

// ---------------------------------------------------------------------------
// Continuous relative position bias table: (169, 6)
// ---------------------------------------------------------------------------
__global__ __launch_bounds__(192) void rpe_kernel(const float* __restrict__ w1,
                                                  const float* __restrict__ b1,
                                                  const float* __restrict__ w2,
                                                  float* __restrict__ table) {
  int idx = threadIdx.x;
  if (idx >= 169) return;
  int i = idx / 13, j = idx % 13;
  auto nf = [](float t) {
    t = t * (8.0f / 6.0f);
    float l = log2f(fabsf(t) + 1.0f) * (1.0f / 3.0f);
    return t < 0.0f ? -l : l;
  };
  float c0 = nf((float)(i - 6));
  float c1 = nf((float)(j - 6));
  float acc[NHh] = {};
  for (int k = 0; k < 512; ++k) {
    float h = fmaxf(c0 * w1[k * 2] + c1 * w1[k * 2 + 1] + b1[k], 0.0f);
#pragma unroll
    for (int hh = 0; hh < NHh; ++hh) acc[hh] += h * w2[hh * 512 + k];
  }
  for (int hh = 0; hh < NHh; ++hh) table[idx * NHh + hh] = acc[hh];
}

// ---------------------------------------------------------------------------
// Fused shifted-window attention: one block per window, one wave per head.
// LDS (halves): [0,36864)  per-head {Q 64x32, K 64x32, Vt 32x64}
//               [36864,50176)  winA 64x208  (reused as O16 64x208)
//               [50176,74752)  per-head P 64x64
// ---------------------------------------------------------------------------
#define LDS_HALVES 74752
__global__ __launch_bounds__(192) void win_attn_kernel(
    const float* __restrict__ x, const _Float16* __restrict__ qkvw16,
    const float* __restrict__ qb, const float* __restrict__ vb,
    const _Float16* __restrict__ projw16, const float* __restrict__ projb,
    const float* __restrict__ table, float* __restrict__ attn_out) {
  extern __shared__ _Float16 sm[];
  const int tid = threadIdx.x;
  const int lane = tid & 31;
  const int wv = tid >> 5;                 // wave id == head id
  const int l15 = lane & 15, lhi = (lane >> 4) & 1;
  const int win = blockIdx.x;
  const int b = win >> 6;
  const int wi = win & 63;
  const int wy = wi >> 3, wx = wi & 7;

  _Float16* qkvLDS = sm;
  _Float16* winA = sm + 36864;             // 64 x 208
  _Float16* Preg = sm + 50176;             // 6 x (64 x 64)

  // ---- phase 1: gather shifted window tile, f32 -> f16, zero-pad rows 49..63
  for (int t = 0; t < 64; ++t) {
    float v = 0.0f;
    if (t < Ntok) {
      int ty = t / WSz, tx = t % WSz;
      int gh = (wy * WSz + ty + SHIFT) % Hh;
      int gw = (wx * WSz + tx + SHIFT) % Ww;
      v = x[((size_t)b * Lseq + gh * Ww + gw) * Cc + tid];
    }
    winA[t * 208 + tid] = (_Float16)v;
  }
  __syncthreads();

  // ---- phase 2: QKV GEMM for this head (cols: which*192 + head*32 + 0..31)
  // 4 independent accumulators (one per M-tile); weight B-fragment hoisted.
  _Float16* Qh  = qkvLDS + wv * 6144;
  _Float16* Kh  = Qh + 2048;
  _Float16* Vth = Qh + 4096;               // [32][64]
  for (int w = 0; w < 3; ++w) {
    for (int dt = 0; dt < 2; ++dt) {
      int j0 = w * Cc + wv * HD + dt * 16;
      int j = j0 + l15;
      float badd = (w == 0) ? qb[j] : ((w == 2) ? vb[j] : 0.0f);
      float mul = (w == 0) ? SCALEQ : 1.0f;
      v8f acc[4] = {};
#pragma unroll
      for (int kk = 0; kk < 6; ++kk) {
        v16h bf = frag_b(qkvw16 + (size_t)j0 * Cc + kk * 32, Cc, lane);
#pragma unroll
        for (int mt = 0; mt < 4; ++mt) {
          v16h a = frag_a(winA + mt * 16 * 208 + kk * 32, 208, lane);
          acc[mt] = wmma32(a, bf, acc[mt]);
        }
      }
      int dloc = dt * 16 + l15;
#pragma unroll
      for (int mt = 0; mt < 4; ++mt) {
        if (w == 2) {
          v8h pk;
#pragma unroll
          for (int v_ = 0; v_ < 8; ++v_) pk[v_] = (_Float16)(acc[mt][v_] + badd);
          *(v8h*)(Vth + dloc * 64 + mt * 16 + lhi * 8) = pk;
        } else {
          _Float16* dst = (w == 0) ? Qh : Kh;
#pragma unroll
          for (int v_ = 0; v_ < 8; ++v_) {
            int m = mt * 16 + v_ + lhi * 8;
            dst[m * HD + dloc] = (_Float16)((acc[mt][v_] + badd) * mul);
          }
        }
      }
    }
  }
  __syncthreads();   // winA fully consumed -> reusable as O16

  // ---- phase 3: attention scores + softmax + P (per head, register softmax)
  _Float16* Ph = Preg + wv * 4096;          // [64][64]
  for (int mt = 0; mt < 4; ++mt) {
    v8f s[4];
    {
      v16h a = frag_a(Qh + mt * 16 * HD, HD, lane);
#pragma unroll
      for (int nt = 0; nt < 4; ++nt) {
        v8f acc = {};
        v16h bf = frag_b(Kh + nt * 16 * HD, HD, lane);
        s[nt] = wmma32(a, bf, acc);
      }
    }
    // bias + shift mask + padding mask
#pragma unroll
    for (int nt = 0; nt < 4; ++nt) {
      int col = nt * 16 + l15;
#pragma unroll
      for (int v_ = 0; v_ < 8; ++v_) {
        int row = mt * 16 + v_ + lhi * 8;
        float val = s[nt][v_];
        if (col >= Ntok) {
          val = -1e30f;
        } else if (row < Ntok) {
          int ty1 = row / WSz, tx1 = row % WSz;
          int ty2 = col / WSz, tx2 = col % WSz;
          int rpi = (ty1 - ty2 + 6) * 13 + (tx1 - tx2 + 6);
          float bias = table[rpi * NHh + wv];
          int p1 = wy * WSz + ty1, q1 = wx * WSz + tx1;
          int p2 = wy * WSz + ty2, q2 = wx * WSz + tx2;
          int r1 = ((p1 < 49) ? 0 : ((p1 < 53) ? 1 : 2)) * 3 +
                   ((q1 < 49) ? 0 : ((q1 < 53) ? 1 : 2));
          int r2 = ((p2 < 49) ? 0 : ((p2 < 53) ? 1 : 2)) * 3 +
                   ((q2 < 49) ? 0 : ((q2 < 53) ? 1 : 2));
          val += bias + ((r1 != r2) ? -100.0f : 0.0f);
        }
        s[nt][v_] = val;
      }
    }
    // row softmax: rows live across the 16-lane half-groups
#pragma unroll
    for (int v_ = 0; v_ < 8; ++v_) {
      float mx = fmaxf(fmaxf(s[0][v_], s[1][v_]), fmaxf(s[2][v_], s[3][v_]));
      mx = fmaxf(mx, __shfl_xor(mx, 1));
      mx = fmaxf(mx, __shfl_xor(mx, 2));
      mx = fmaxf(mx, __shfl_xor(mx, 4));
      mx = fmaxf(mx, __shfl_xor(mx, 8));
      float e0 = __expf(s[0][v_] - mx), e1 = __expf(s[1][v_] - mx);
      float e2 = __expf(s[2][v_] - mx), e3 = __expf(s[3][v_] - mx);
      float sum = e0 + e1 + e2 + e3;
      sum += __shfl_xor(sum, 1);
      sum += __shfl_xor(sum, 2);
      sum += __shfl_xor(sum, 4);
      sum += __shfl_xor(sum, 8);
      float inv = 1.0f / sum;
      int row = mt * 16 + v_ + lhi * 8;
      Ph[row * 64 +  0 + l15] = (_Float16)(e0 * inv);
      Ph[row * 64 + 16 + l15] = (_Float16)(e1 * inv);
      Ph[row * 64 + 32 + l15] = (_Float16)(e2 * inv);
      Ph[row * 64 + 48 + l15] = (_Float16)(e3 * inv);
    }
  }
  asm volatile("s_wait_dscnt 0" ::: "memory");

  // ---- phase 4: out = P @ V, store into O16 (winA region) per head
  // shared A-fragment, 2 independent accumulators per M-tile
  _Float16* O16 = winA;                    // 64 x 208, cols wv*32..wv*32+31
  for (int mt = 0; mt < 4; ++mt) {
    v8f acc[2] = {};
#pragma unroll
    for (int kt = 0; kt < 2; ++kt) {
      v16h a = frag_a(Ph + mt * 16 * 64 + kt * 32, 64, lane);
#pragma unroll
      for (int dt = 0; dt < 2; ++dt) {
        v16h bf = frag_b(Vth + dt * 16 * 64 + kt * 32, 64, lane);
        acc[dt] = wmma32(a, bf, acc[dt]);
      }
    }
#pragma unroll
    for (int dt = 0; dt < 2; ++dt) {
      int j = wv * HD + dt * 16 + l15;
#pragma unroll
      for (int v_ = 0; v_ < 8; ++v_) {
        int m = mt * 16 + v_ + lhi * 8;
        O16[m * 208 + j] = (_Float16)acc[dt][v_];
      }
    }
  }
  __syncthreads();

  // ---- phase 5: projection (49x192 @ 192x192) + window-reverse scatter store
  // 4 independent accumulators, weight B-fragment hoisted
  for (int ns = 0; ns < 2; ++ns) {
    int j0 = wv * 32 + ns * 16;
    int j = j0 + l15;
    float pb = projb[j];
    v8f acc[4] = {};
#pragma unroll
    for (int kk = 0; kk < 6; ++kk) {
      v16h bf = frag_b(projw16 + (size_t)j0 * Cc + kk * 32, Cc, lane);
#pragma unroll
      for (int mt = 0; mt < 4; ++mt) {
        v16h a = frag_a(O16 + mt * 16 * 208 + kk * 32, 208, lane);
        acc[mt] = wmma32(a, bf, acc[mt]);
      }
    }
#pragma unroll
    for (int mt = 0; mt < 4; ++mt) {
#pragma unroll
      for (int v_ = 0; v_ < 8; ++v_) {
        int m = mt * 16 + v_ + lhi * 8;
        if (m < Ntok) {
          int ty = m / WSz, tx = m % WSz;
          int gh = (wy * WSz + ty + SHIFT) % Hh;
          int gw = (wx * WSz + tx + SHIFT) % Ww;
          attn_out[((size_t)b * Lseq + gh * Ww + gw) * Cc + j] = acc[mt][v_] + pb;
        }
      }
    }
  }
}

// ---------------------------------------------------------------------------
// LayerNorm(src) * w + b + resid -> out32 (and optional f16 copy)
// one wave per row of 192
// ---------------------------------------------------------------------------
__global__ __launch_bounds__(256) void ln_res_kernel(
    const float* __restrict__ src, const float* __restrict__ resid,
    const float* __restrict__ w, const float* __restrict__ bsh,
    float* __restrict__ out32, _Float16* __restrict__ out16, int rows) {
  int lane = threadIdx.x & 31;
  int wv = threadIdx.x >> 5;
  int row = blockIdx.x * 8 + wv;
  if (row >= rows) return;
  const float* r = src + (size_t)row * Cc;
  float v[6];
  float sum = 0.0f, sq = 0.0f;
#pragma unroll
  for (int i = 0; i < 6; ++i) {
    v[i] = r[lane + i * 32];
    sum += v[i];
    sq += v[i] * v[i];
  }
#pragma unroll
  for (int mk = 1; mk < 32; mk <<= 1) {
    sum += __shfl_xor(sum, mk);
    sq += __shfl_xor(sq, mk);
  }
  float mean = sum * (1.0f / Cc);
  float var = sq * (1.0f / Cc) - mean * mean;
  float rs = rsqrtf(var + 1e-5f);
#pragma unroll
  for (int i = 0; i < 6; ++i) {
    int c = lane + i * 32;
    float y = (v[i] - mean) * rs * w[c] + bsh[c] + resid[(size_t)row * Cc + c];
    out32[(size_t)row * Cc + c] = y;
    if (out16) out16[(size_t)row * Cc + c] = (_Float16)y;
  }
}

// ---------------------------------------------------------------------------
// C(M,N) = act(A(MxK,f16) @ Wt(NxK,f16)^T + bias); mode 0: GELU->f16, 1: f32
// block = 128 threads (4 waves), block tile 64(M) x 64(N)
// 4 independent N-tile accumulators share each A-fragment; A stream prefetched.
// ---------------------------------------------------------------------------
__global__ __launch_bounds__(128) void gemm_kernel(
    const _Float16* __restrict__ A, const _Float16* __restrict__ Wt,
    const float* __restrict__ bias, void* __restrict__ out,
    int M, int N, int K, int mode) {
  const int lane = threadIdx.x & 31;
  const int wv = threadIdx.x >> 5;
  const int l15 = lane & 15, lhi = (lane >> 4) & 1;
  const int m0 = blockIdx.y * 64 + wv * 16;
  const int nbase = blockIdx.x * 64;
  const _Float16* pa = A + (size_t)m0 * K;
  v8f acc[4] = {};
  for (int k0 = 0; k0 < K; k0 += 32) {
    v16h a = frag_a(pa + k0, K, lane);
    if (k0 + 32 < K) {
      // next A chunk for this lane -> global_prefetch_b8 (A is streamed once;
      // weight tiles stay hot in the 192MB L2 across the 1568 M-blocks)
      __builtin_prefetch((const void*)(pa + (size_t)l15 * K + k0 + 32 + lhi * 8), 0, 1);
    }
#pragma unroll
    for (int nt = 0; nt < 4; ++nt) {
      v16h b = frag_b(Wt + (size_t)(nbase + nt * 16) * K + k0, K, lane);
      acc[nt] = wmma32(a, b, acc[nt]);
    }
  }
#pragma unroll
  for (int nt = 0; nt < 4; ++nt) {
    int n = nbase + nt * 16 + l15;
    float bv = bias[n];
    if (mode == 0) {
      _Float16* o = (_Float16*)out;
#pragma unroll
      for (int v_ = 0; v_ < 8; ++v_) {
        int m = m0 + v_ + lhi * 8;
        float xv = acc[nt][v_] + bv;
        float g = 0.5f * xv * (1.0f + erff(xv * 0.7071067811865475f));
        o[(size_t)m * N + n] = (_Float16)g;
      }
    } else {
      float* o = (float*)out;
#pragma unroll
      for (int v_ = 0; v_ < 8; ++v_) {
        int m = m0 + v_ + lhi * 8;
        o[(size_t)m * N + n] = acc[nt][v_] + bv;
      }
    }
  }
}

// ---------------------------------------------------------------------------
// Host launcher
// ---------------------------------------------------------------------------
extern "C" void kernel_launch(void* const* d_in, const int* in_sizes, int n_in,
                              void* d_out, int out_size, void* d_ws, size_t ws_size,
                              hipStream_t stream) {
  (void)in_sizes; (void)n_in; (void)out_size; (void)ws_size;
  const float* x       = (const float*)d_in[0];
  const float* qkv_w   = (const float*)d_in[1];
  const float* q_bias  = (const float*)d_in[2];
  const float* v_bias  = (const float*)d_in[3];
  const float* proj_w  = (const float*)d_in[4];
  const float* proj_b  = (const float*)d_in[5];
  const float* rpe_w1  = (const float*)d_in[6];
  const float* rpe_b1  = (const float*)d_in[7];
  const float* rpe_w2  = (const float*)d_in[8];
  const float* norm1_w = (const float*)d_in[9];
  const float* norm1_b = (const float*)d_in[10];
  const float* norm2_w = (const float*)d_in[11];
  const float* norm2_b = (const float*)d_in[12];
  const float* fc1_w   = (const float*)d_in[13];
  const float* fc1_b   = (const float*)d_in[14];
  const float* fc2_w   = (const float*)d_in[15];
  const float* fc2_b   = (const float*)d_in[16];

  char* ws = (char*)d_ws;
  size_t off = 0;
  auto take = [&](size_t bytes) {
    char* p = ws + off;
    off += (bytes + 255) & ~(size_t)255;
    return p;
  };
  _Float16* qkvw16  = (_Float16*)take((size_t)576 * 192 * 2);
  _Float16* projw16 = (_Float16*)take((size_t)192 * 192 * 2);
  _Float16* fc1w16  = (_Float16*)take((size_t)768 * 192 * 2);
  _Float16* fc2w16  = (_Float16*)take((size_t)192 * 768 * 2);
  float*    table   = (float*)take((size_t)169 * 6 * 4);
  float*    attnout = (float*)take((size_t)ROWS * Cc * 4);
  float*    x1f32   = (float*)take((size_t)ROWS * Cc * 4);
  _Float16* x1f16   = (_Float16*)take((size_t)ROWS * Cc * 2);
  _Float16* h16     = (_Float16*)take((size_t)ROWS * 768 * 2);
  float*    h2f32   = (float*)take((size_t)ROWS * Cc * 4);

  cvt_kernel<<<(576 * 192 + 255) / 256, 256, 0, stream>>>(qkv_w, qkvw16, 576 * 192);
  cvt_kernel<<<(192 * 192 + 255) / 256, 256, 0, stream>>>(proj_w, projw16, 192 * 192);
  cvt_kernel<<<(768 * 192 + 255) / 256, 256, 0, stream>>>(fc1_w, fc1w16, 768 * 192);
  cvt_kernel<<<(192 * 768 + 255) / 256, 256, 0, stream>>>(fc2_w, fc2w16, 192 * 768);
  rpe_kernel<<<1, 192, 0, stream>>>(rpe_w1, rpe_b1, rpe_w2, table);

  const size_t SMEM = (size_t)LDS_HALVES * sizeof(_Float16);  // 149504 B (<320KB/WGP)
  (void)hipFuncSetAttribute((const void*)win_attn_kernel,
                            hipFuncAttributeMaxDynamicSharedMemorySize, (int)SMEM);
  win_attn_kernel<<<Bb * NWIN, 192, SMEM, stream>>>(x, qkvw16, q_bias, v_bias,
                                                    projw16, proj_b, table, attnout);

  ln_res_kernel<<<ROWS / 8, 256, 0, stream>>>(attnout, x, norm1_w, norm1_b,
                                              x1f32, x1f16, ROWS);
  gemm_kernel<<<dim3(768 / 64, ROWS / 64), 128, 0, stream>>>(
      x1f16, fc1w16, fc1_b, (void*)h16, ROWS, 768, Cc, 0);
  gemm_kernel<<<dim3(192 / 64, ROWS / 64), 128, 0, stream>>>(
      h16, fc2w16, fc2_b, (void*)h2f32, ROWS, Cc, 768, 1);
  ln_res_kernel<<<ROWS / 8, 256, 0, stream>>>(h2f32, x1f32, norm2_w, norm2_b,
                                              (float*)d_out, (_Float16*)nullptr, ROWS);
}
